// VectorQuantizer_32873679683613
// MI455X (gfx1250) — matrix operand order repaired and checked by reading, not compiled, not run on hardware
//
#include <hip/hip_runtime.h>

// ---------------------------------------------------------------------------
// VQ-VAE quantizer for MI455X (gfx1250), wave32.
// Distance matmul done with V_WMMA_F32_16X16X32_BF16 using an error-
// compensated bf16 split (hi+lo): S = Ah*Bh + Ah*Bl + Al*Bh  (fp32 accum).
// argmin gap ~1e-4 vs split error ~1e-7 -> exact argmin, ~8x the FLOP rate
// of the fp32 16x16x4 WMMA path.
// ---------------------------------------------------------------------------

typedef __attribute__((ext_vector_type(16))) __bf16 v16bf;
typedef __attribute__((ext_vector_type(8)))  float  v8f;
typedef __attribute__((ext_vector_type(2)))  float  v2f;

#define SUPER   256                   // codes staged per super-chunk
#define NSUPER  4
#define CDIM    64
#define HW      4096                  // 64*64 pixels per (b,c) plane
#define NTOT    (32 * 64 * 64 * 64)   // 8388608 output elements

// LDS B-operand slots, 16B each, operand-ordered:
//   flat = kb*256 + step*128 + term*64 + piece*32 + lane
#define SLOT(kb, s, t, p, ln) ((kb)*256 + (s)*128 + (t)*64 + (p)*32 + (ln))

struct BPair { int4 a, b; };          // 32B, feeds one v16bf via bit_cast

static __device__ __forceinline__ void bsplit(float x, __bf16& h, __bf16& l) {
  h = (__bf16)x;                      // RNE
  l = (__bf16)(x - (float)h);         // residual exact in f32
}

static __device__ __forceinline__ unsigned pk2(__bf16 a, __bf16 b) {
  union { __bf16 h[2]; unsigned u; } r;
  r.h[0] = a; r.h[1] = b;             // low half = even K, high half = odd K
  return r.u;
}

static __device__ __forceinline__ v8f wmma_bf16(v16bf a, v16bf b, v8f c) {
  return __builtin_amdgcn_wmma_f32_16x16x32_bf16(
      false, a, false, b, (short)0, c, false, false);
}

__global__ void vq_zero_loss(float* loss) {
  if (threadIdx.x == 0) *loss = 0.0f;
}

__global__ __launch_bounds__(256)
void vq_main(const float* __restrict__ z,
             const float* __restrict__ Eg,
             float* __restrict__ out,
             float* __restrict__ loss) {
  __shared__ int4  bq[4096];          // 64KB: split codebook, operand order
  __shared__ float enorm[SUPER];      // ||e_k||^2 (fp32, exact)
  __shared__ int   kminLds[128];      // per-pixel argmin of this workgroup

  const int tid  = threadIdx.x;
  const int lane = tid & 31;
  const int wave = tid >> 5;
  const int li   = lane & 15;         // A-row M / D-col N
  const int hi   = lane >> 4;         // K-half (A/B) and M-half (D)

  const int blk = blockIdx.x;         // 1024 blocks
  const int b   = blk >> 5;
  const int p0  = ((blk & 31) << 7) + (wave << 4);   // wave pixel base

  // ---- A tile: 16 pixels x 64 ch; bf16 split, fp32 copy kept for loss ----
  // element i of step s maps to channel c = 32s + 16*(i>>3) + 8*hi + (i&7)
  const float* zb = z + (size_t)b * (CDIM * HW) + (p0 + li);
  v16bf AH[2], AL[2];
  float zf[32];                       // (s*2+g)*8 + o
#pragma unroll
  for (int s = 0; s < 2; ++s) {
#pragma unroll
    for (int g = 0; g < 2; ++g) {
      const int c0 = 32 * s + 16 * g + 8 * hi;
#pragma unroll
      for (int o = 0; o < 8; ++o) {
        const float x = zb[(size_t)(c0 + o) * HW];
        zf[(s * 2 + g) * 8 + o] = x;
        __bf16 h, l;
        bsplit(x, h, l);
        AH[s][8 * g + o] = h;
        AL[s][8 * g + o] = l;
      }
    }
  }

  float minv[8];
  int   mini[8];
#pragma unroll
  for (int v = 0; v < 8; ++v) { minv[v] = 3.0e38f; mini[v] = 0; }

  for (int s = 0; s < NSUPER; ++s) {
    __syncthreads();
    // ---- stage 256 codes: bf16-split, packed into WMMA-operand slots ----
    {
      const int kk = tid;             // one code per thread
      const float4* src4 = (const float4*)(Eg + (size_t)(s * SUPER + kk) * CDIM);
      float nrm = 0.0f;
      const int kb = kk >> 4, kl = kk & 15;
#pragma unroll
      for (int g8 = 0; g8 < 8; ++g8) {          // channel octet c0 = 8*g8
        const float4 fa = src4[2 * g8];
        const float4 fb = src4[2 * g8 + 1];
        nrm += fa.x * fa.x + fa.y * fa.y + fa.z * fa.z + fa.w * fa.w
             + fb.x * fb.x + fb.y * fb.y + fb.z * fb.z + fb.w * fb.w;
        const float f[8] = {fa.x, fa.y, fa.z, fa.w, fb.x, fb.y, fb.z, fb.w};
        unsigned h[4], lo[4];
#pragma unroll
        for (int j = 0; j < 4; ++j) {
          __bf16 h0, l0, h1, l1;
          bsplit(f[2 * j], h0, l0);
          bsplit(f[2 * j + 1], h1, l1);
          h[j]  = pk2(h0, h1);
          lo[j] = pk2(l0, l1);
        }
        const int st = g8 >> 2, hf = g8 & 1, pc = (g8 >> 1) & 1;
        const int ln = kl + 16 * hf;
        bq[SLOT(kb, st, 0, pc, ln)] = make_int4((int)h[0], (int)h[1], (int)h[2], (int)h[3]);
        bq[SLOT(kb, st, 1, pc, ln)] = make_int4((int)lo[0], (int)lo[1], (int)lo[2], (int)lo[3]);
      }
      enorm[kk] = nrm;
    }
    __syncthreads();

    // ---- 16 chunks of 16 codes: 6 bf16 WMMAs per chunk (hi*hi+hi*lo+lo*hi)
    for (int kb = 0; kb < 16; ++kb) {
      BPair ph0, ph1, pl0, pl1;       // conflict-free ds_load_b128 pairs
      ph0.a = bq[SLOT(kb, 0, 0, 0, lane)];
      ph0.b = bq[SLOT(kb, 0, 0, 1, lane)];
      ph1.a = bq[SLOT(kb, 1, 0, 0, lane)];
      ph1.b = bq[SLOT(kb, 1, 0, 1, lane)];
      pl0.a = bq[SLOT(kb, 0, 1, 0, lane)];
      pl0.b = bq[SLOT(kb, 0, 1, 1, lane)];
      pl1.a = bq[SLOT(kb, 1, 1, 0, lane)];
      pl1.b = bq[SLOT(kb, 1, 1, 1, lane)];
      const v16bf BH0 = __builtin_bit_cast(v16bf, ph0);
      const v16bf BH1 = __builtin_bit_cast(v16bf, ph1);
      const v16bf BL0 = __builtin_bit_cast(v16bf, pl0);
      const v16bf BL1 = __builtin_bit_cast(v16bf, pl1);

      const float en = enorm[kb * 16 + li];
      v8f acc0 = {};
      v8f acc1 = {};
      acc0 = wmma_bf16(AH[0], BH0, acc0);
      acc1 = wmma_bf16(AH[1], BH1, acc1);
      acc0 = wmma_bf16(AH[0], BL0, acc0);
      acc1 = wmma_bf16(AH[1], BL1, acc1);
      acc0 = wmma_bf16(AL[0], BH0, acc0);
      acc1 = wmma_bf16(AL[1], BH1, acc1);

      const int kg = s * SUPER + kb * 16 + li;  // global code index
#pragma unroll
      for (int v = 0; v < 8; ++v) {             // pixel m = v + 8*hi
        const float d = en - 2.0f * (acc0[v] + acc1[v]);
        if (d < minv[v]) { minv[v] = d; mini[v] = kg; }
      }
    }
  }

  // ---- argmin butterfly within each 16-lane half (ties -> smaller index) --
#pragma unroll
  for (int v = 0; v < 8; ++v) {
    float mv = minv[v];
    int   mk = mini[v];
#pragma unroll
    for (int off = 1; off <= 8; off <<= 1) {
      const float ov = __shfl_xor(mv, off, 32);
      const int   ok = __shfl_xor(mk, off, 32);
      if (ov < mv || (ov == mv && ok < mk)) { mv = ov; mk = ok; }
    }
    minv[v] = mv;
    mini[v] = mk;
  }
  if (li == 0) {
#pragma unroll
    for (int v = 0; v < 8; ++v)
      kminLds[(wave << 4) + hi * 8 + v] = mini[v];
  }
  __syncthreads();
  const int kmin = kminLds[(wave << 4) + li];   // argmin for pixel p0+li

  // ---- gather winning embedding, write out[B,C,H,W], fused exact loss ----
  const float* erow = Eg + (size_t)kmin * CDIM;
  float* ob = out + (size_t)b * (CDIM * HW) + (p0 + li);
  float lsum = 0.0f;
#pragma unroll
  for (int s = 0; s < 2; ++s) {
#pragma unroll
    for (int g = 0; g < 2; ++g) {
      const int c0 = 32 * s + 16 * g + 8 * hi;
#pragma unroll
      for (int o2 = 0; o2 < 4; ++o2) {
        const int c = c0 + 2 * o2;
        const v2f qv = *(const v2f*)(erow + c);     // L2-resident b64 gather
        const float dx = qv.x - zf[(s * 2 + g) * 8 + 2 * o2];
        const float dy = qv.y - zf[(s * 2 + g) * 8 + 2 * o2 + 1];
        lsum += dx * dx + dy * dy;
        ob[(size_t)c * HW]       = qv.x;            // 64B-coalesced stores
        ob[(size_t)(c + 1) * HW] = qv.y;
      }
    }
  }
#pragma unroll
  for (int off = 1; off < 32; off <<= 1) lsum += __shfl_xor(lsum, off, 32);
  if (lane == 0) atomicAdd(loss, lsum * (1.25f / (float)NTOT));
}

extern "C" void kernel_launch(void* const* d_in, const int* in_sizes, int n_in,
                              void* d_out, int out_size, void* d_ws, size_t ws_size,
                              hipStream_t stream) {
  (void)in_sizes; (void)n_in; (void)out_size; (void)d_ws; (void)ws_size;
  const float* z  = (const float*)d_in[0];   // [32, 64, 64, 64]
  const float* Eg = (const float*)d_in[1];   // [1024, 64]
  float* out  = (float*)d_out;               // [32, 64, 64, 64] flattened
  float* loss = out + NTOT;                  // scalar appended after out

  vq_zero_loss<<<1, 32, 0, stream>>>(loss);
  vq_main<<<1024, 256, 0, stream>>>(z, Eg, out, loss);
}
